// RingDilatedAttentionHilbertOptimizedFixed_68959994905216
// MI455X (gfx1250) — matrix operand order repaired and checked by reading, not compiled, not run on hardware
//
#include <hip/hip_runtime.h>

typedef __attribute__((ext_vector_type(16))) _Float16 v16h;
typedef __attribute__((ext_vector_type(8)))  float    v8f;
typedef __attribute__((ext_vector_type(4))) unsigned int v4u;
typedef __attribute__((ext_vector_type(8)))  int      v8i;
typedef __attribute__((ext_vector_type(4)))  int      v4i;

#define NSEQ 4096
#define NHEAD 16
#define DH 64
#define BQ 128
#define BK 64
#define NKT (2048 / BK)
#define STR 72          // padded f16 row stride produced by TDM pad (32dw + 4dw)
#define NSLOT 48        // (3 problems x 2 batches) x 8 heads

// LDS element offsets within the dynamic shared block
#define QS_E 0
#define KT_E (BQ * STR)                    // 2 buffers of BK*STR
#define VS_E (BQ * STR + 2 * BK * STR)     // 2 buffers of BK*STR
#define PS_E (BQ * STR + 4 * BK * STR)     // 8 waves x 16 x STR

// ---------------- Hilbert permutation: perm[dist] = y*64 + x ----------------
__global__ void hilbert_perm_kernel(int* perm) {
    int d = blockIdx.x * blockDim.x + threadIdx.x;
    if (d >= NSEQ) return;
    int t = d, x = 0, y = 0;
    #pragma unroll
    for (int s = 1; s < 64; s <<= 1) {
        int rx = 1 & (t >> 1);
        int ry = 1 & (t ^ rx);
        if (ry == 0) {
            if (rx == 1) { x = s - 1 - x; y = s - 1 - y; }
            int tmp = x; x = y; y = tmp;
        }
        x += s * rx;
        y += s * ry;
        t >>= 2;
    }
    perm[d] = y * 64 + x;
}

// gather tables: p0 = perm[j], p1 = perm[2048+j], p2 = perm[1+2j]
__global__ void gather_idx_kernel(const int* __restrict__ perm, int* __restrict__ gidx) {
    int j = blockIdx.x * blockDim.x + threadIdx.x;
    if (j >= 2048) return;
    gidx[j]        = perm[j];
    gidx[2048 + j] = perm[2048 + j];
    gidx[4096 + j] = perm[1 + 2 * j];
}

__global__ void zero_out_kernel(float4* out, int n4) {
    int i = blockIdx.x * blockDim.x + threadIdx.x;
    if (i < n4) out[i] = make_float4(0.f, 0.f, 0.f, 0.f);
}

// ---------------- gather + f32->f16 convert pre-pass (done ONCE per slot) ----
// Qg/Vg: [slot][2048][64] row-major.  Ktg: [slot][64][2048] (pre-transposed).
// grid = (2048/256, 48), block = 256
__global__ __launch_bounds__(256) void gather_cvt_kernel(
        const float* __restrict__ q, const float* __restrict__ k,
        const float* __restrict__ v, const int* __restrict__ gidx,
        _Float16* __restrict__ Qg, _Float16* __restrict__ Ktg,
        _Float16* __restrict__ Vg) {
    int slot = blockIdx.y;                    // (p*2+b)*8 + hp
    int p  = slot >> 4;
    int bb = (slot >> 3) & 1;
    int hp = slot & 7;
    int hg = (p == 2) ? 8 + hp : hp;
    int j = blockIdx.x * blockDim.x + threadIdx.x;   // 0..2047
    int nsrc = gidx[p * 2048 + j];
    size_t src    = ((size_t)(bb * NSEQ + nsrc) * NHEAD + hg) * DH;
    size_t rowdst = ((size_t)slot * 2048 + j) * DH;
    size_t ktb    = (size_t)slot * DH * 2048;
    #pragma unroll
    for (int c0 = 0; c0 < DH; c0 += 16) {
        _Float16 tq[16], tv[16];
        #pragma unroll
        for (int c = 0; c < 16; ++c) {
            tq[c] = (_Float16)(q[src + c0 + c] * 0.125f);   // fold 1/sqrt(D)
            tv[c] = (_Float16)(v[src + c0 + c]);
            Ktg[ktb + (size_t)(c0 + c) * 2048 + j] = (_Float16)(k[src + c0 + c]);
        }
        v4u* dq = (v4u*)(Qg + rowdst + c0);
        v4u* dv = (v4u*)(Vg + rowdst + c0);
        dq[0] = ((v4u*)tq)[0]; dq[1] = ((v4u*)tq)[1];
        dv[0] = ((v4u*)tv)[0]; dv[1] = ((v4u*)tv)[1];
    }
}

// ---------------- TDM 2D tile load: global f16 -> padded LDS rows ------------
// tile_w elems per row (x2B = multiple of 4B), tile_h rows, row stride in elems.
// Pad: every 32 DWORDs (64 halves) insert 4 DWORDs (8 halves) -> LDS stride 72.
__device__ inline void tdm_load_2d(unsigned lds_off, const _Float16* gaddr,
                                   unsigned tile_w, unsigned tile_h,
                                   unsigned tensor_w, unsigned tensor_h,
                                   unsigned row_stride) {
    unsigned long long ga = (unsigned long long)gaddr;
    v4u g0;
    g0[0] = 1u;                                    // count=1, user D#
    g0[1] = lds_off;                               // LDS byte address
    g0[2] = (unsigned)ga;                          // global_addr[31:0]
    g0[3] = (unsigned)(ga >> 32) | (2u << 30);     // global_addr[56:32] | type=2
    v8i g1;
    g1[0] = (int)((1u << 16)       // data_size = 2 bytes
                | (1u << 20)       // pad_enable
                | (4u << 22)       // pad_interval: 32 DWORDs
                | (3u << 25));     // pad_amount:   4 DWORDs
    g1[1] = (int)((tensor_w & 0xFFFFu) << 16);                    // dim0[15:0]
    g1[2] = (int)((tensor_w >> 16) | ((tensor_h & 0xFFFFu) << 16));
    g1[3] = (int)((tensor_h >> 16) | (tile_w << 16));             // tile_dim0
    g1[4] = (int)(tile_h & 0xFFFFu);                              // tile_dim1
    g1[5] = (int)row_stride;                                      // dim0_stride
    g1[6] = 0;
    g1[7] = 0;
    v4i z4 = {0, 0, 0, 0};
    v8i z8 = {0, 0, 0, 0, 0, 0, 0, 0};
    __builtin_amdgcn_tensor_load_to_lds(g0, g1, z4, z4, z8, 0);
}

// ---------------- vectorized fragment loaders (ISA 7.12.2 16-bit layouts) ----
__device__ inline v16h ld_frag_2x16B(const _Float16* p0, const _Float16* p1) {
    union { v4u u[2]; v16h h; } t;
    t.u[0] = *(const v4u*)p0;
    t.u[1] = *(const v4u*)p1;
    return t.h;
}

__device__ inline v16h load_a_frag(const _Float16* base, int row0, int k0, int lane) {
    int m = lane & 15, h = lane >> 4;
    const _Float16* p = base + (row0 + m) * STR + k0 + 8 * h;
    return ld_frag_2x16B(p, p + 16);
}

__device__ inline v16h load_b_frag(const _Float16* base, int k0, int n0, int lane) {
    int n = lane & 15, h = lane >> 4;
    const _Float16* p = base + (k0 + 8 * h) * STR + n0 + n;
    v16h r;
    #pragma unroll
    for (int s = 0; s < 8; ++s) {
        r[s]     = p[s * STR];
        r[s + 8] = p[(16 + s) * STR];
    }
    return r;
}

#define WMMA_F16(A, B, C) \
    __builtin_amdgcn_wmma_f32_16x16x32_f16(false, (A), false, (B), (short)0, (C), false, false)

// 16-lane circular max reduction in pure VALU (DPP16 row_ror 1/2/4/8)
__device__ inline float row16_max(float x) {
    x = fmaxf(x, __int_as_float(__builtin_amdgcn_update_dpp(
            0, __float_as_int(x), 0x121, 0xf, 0xf, true)));
    x = fmaxf(x, __int_as_float(__builtin_amdgcn_update_dpp(
            0, __float_as_int(x), 0x122, 0xf, 0xf, true)));
    x = fmaxf(x, __int_as_float(__builtin_amdgcn_update_dpp(
            0, __float_as_int(x), 0x124, 0xf, 0xf, true)));
    x = fmaxf(x, __int_as_float(__builtin_amdgcn_update_dpp(
            0, __float_as_int(x), 0x128, 0xf, 0xf, true)));
    return x;
}

// ---------------- flash attention, TDM-staged, double-buffered ---------------
// grid = (16 qblocks, 3 problems, 2*8); block = 256 (8 waves)
// dynamic LDS: Qs[128*72] Kt0 Kt1 Vs0 Vs1 [64*72 each] Ps[8*16*72]  = 73728 B
__global__ __launch_bounds__(256) void dilated_flash_attn_kernel(
        const _Float16* __restrict__ Qg, const _Float16* __restrict__ Ktg,
        const _Float16* __restrict__ Vg, const int* __restrict__ gidx,
        float* __restrict__ out) {
    extern __shared__ _Float16 smem[];
    _Float16* Qs = smem + QS_E;
    _Float16* Ps = smem + PS_E;

    const int qb   = blockIdx.x;
    const int prob = blockIdx.y;
    const int bz   = blockIdx.z;
    const int b    = bz >> 3;
    const int head = bz & 7;
    const int hg   = (prob == 2) ? (8 + head) : head;
    const int slot = (prob * 2 + b) * 8 + head;
    const int* gp  = gidx + prob * 2048;

    const int tid  = threadIdx.x;
    const int lane = tid & 31;
    const int w    = tid >> 5;

    // LDS aperture: low 32 bits of a flat LDS pointer == LDS byte offset.
    // Runtime scalar; no pointer arrays (avoids addrspacecast in initializers).
    const unsigned base = (unsigned)(size_t)(void*)smem;

    const _Float16* qbase  = Qg  + (size_t)slot * 2048 * DH + (size_t)qb * BQ * DH;
    const _Float16* ktbase = Ktg + (size_t)slot * DH * 2048;
    const _Float16* vbase  = Vg  + (size_t)slot * 2048 * DH;

    // prologue: DMA Q tile + first K/V tiles, then publish
    if (w == 0) {
        tdm_load_2d(base + QS_E * 2, qbase,  DH, BQ, DH, 2048, DH);     // 128x64
        tdm_load_2d(base + KT_E * 2, ktbase, BK, DH, 2048, DH, 2048);   // 64 rows of Kt
        tdm_load_2d(base + VS_E * 2, vbase,  DH, BK, DH, 2048, DH);     // 64x64
        __builtin_amdgcn_s_wait_tensorcnt(0);
    }
    __syncthreads();

    v16h ones;
    #pragma unroll
    for (int e = 0; e < 16; ++e) ones[e] = (_Float16)1.0f;

    const v16h qa0 = load_a_frag(Qs, w * 16, 0,  lane);
    const v16h qa1 = load_a_frag(Qs, w * 16, 32, lane);

    v8f o0 = {}, o1 = {}, o2 = {}, o3 = {};
    float mi[8], li[8];
    #pragma unroll
    for (int r = 0; r < 8; ++r) { mi[r] = -1e30f; li[r] = 0.f; }

    for (int kb = 0; kb < NKT; ++kb) {
        const int cur = kb & 1, nxt = cur ^ 1;
        const _Float16* Kt = smem + KT_E + cur * (BK * STR);
        const _Float16* Vs = smem + VS_E + cur * (BK * STR);

        // kick off next tile's DMA into the other buffer (runs under compute)
        if (w == 0 && kb + 1 < NKT) {
            tdm_load_2d(base + (KT_E + nxt * (BK * STR)) * 2,
                        ktbase + (kb + 1) * BK, BK, DH, 2048, DH, 2048);
            tdm_load_2d(base + (VS_E + nxt * (BK * STR)) * 2,
                        vbase + (size_t)(kb + 1) * BK * DH, DH, BK, DH, 2048, DH);
        }

        // ---- S = Q * K^T  (8 WMMAs; B from Kt is contiguous) ----
        v8f s[4];
        #pragma unroll
        for (int nt = 0; nt < 4; ++nt) {
            v8f acc = {};
            acc = WMMA_F16(qa0, load_b_frag(Kt, 0,  nt * 16, lane), acc);
            acc = WMMA_F16(qa1, load_b_frag(Kt, 32, nt * 16, lane), acc);
            s[nt] = acc;
        }

        // ---- online softmax: DPP max butterfly, exp in TRANS pipe ----
        float alpha[8];
        #pragma unroll
        for (int r = 0; r < 8; ++r) {
            float mx = row16_max(fmaxf(fmaxf(s[0][r], s[1][r]), fmaxf(s[2][r], s[3][r])));
            float mnew = fmaxf(mi[r], mx);
            alpha[r] = __expf(mi[r] - mnew);
            mi[r] = mnew;
            #pragma unroll
            for (int nt = 0; nt < 4; ++nt)
                s[nt][r] = __expf(s[nt][r] - mnew);
        }

        // ---- P -> per-wave LDS (C-frag -> A-frag relayout), rescale O ----
        _Float16* pw = Ps + w * 16 * STR;
        {
            int n = lane & 15, h = lane >> 4;
            #pragma unroll
            for (int nt = 0; nt < 4; ++nt)
                #pragma unroll
                for (int r = 0; r < 8; ++r)
                    pw[(8 * h + r) * STR + nt * 16 + n] = (_Float16)s[nt][r];
        }
        #pragma unroll
        for (int r = 0; r < 8; ++r) {
            o0[r] *= alpha[r]; o1[r] *= alpha[r];
            o2[r] *= alpha[r]; o3[r] *= alpha[r];
        }
        __asm__ volatile("s_wait_dscnt 0" ::: "memory");  // wave-local LDS RAW

        // ---- O += P * V, rowsum(P) on the matrix pipe (10 WMMAs) ----
        const v16h pa0 = load_a_frag(pw, 0, 0,  lane);
        const v16h pa1 = load_a_frag(pw, 0, 32, lane);
        v8f rsum = {};
        rsum = WMMA_F16(pa0, ones, rsum);
        rsum = WMMA_F16(pa1, ones, rsum);
        o0 = WMMA_F16(pa0, load_b_frag(Vs, 0,  0,  lane), o0);
        o0 = WMMA_F16(pa1, load_b_frag(Vs, 32, 0,  lane), o0);
        o1 = WMMA_F16(pa0, load_b_frag(Vs, 0,  16, lane), o1);
        o1 = WMMA_F16(pa1, load_b_frag(Vs, 32, 16, lane), o1);
        o2 = WMMA_F16(pa0, load_b_frag(Vs, 0,  32, lane), o2);
        o2 = WMMA_F16(pa1, load_b_frag(Vs, 32, 32, lane), o2);
        o3 = WMMA_F16(pa0, load_b_frag(Vs, 0,  48, lane), o3);
        o3 = WMMA_F16(pa1, load_b_frag(Vs, 32, 48, lane), o3);
        #pragma unroll
        for (int r = 0; r < 8; ++r)
            li[r] = li[r] * alpha[r] + rsum[r];

        // publish next buffer: wave0 waits DMA done, barrier orders everyone
        if (w == 0) __builtin_amdgcn_s_wait_tensorcnt(0);
        __syncthreads();
    }

    // ---- epilogue: normalize and scatter through gather index ----
    {
        int n = lane & 15, h = lane >> 4;
        #pragma unroll
        for (int r = 0; r < 8; ++r) {
            int m = 8 * h + r;
            int j = qb * BQ + w * 16 + m;
            int nsrc = gp[j];
            float inv_l = 1.0f / li[r];
            float* dst = out + (((size_t)(b * NSEQ + nsrc) * NHEAD + hg) * DH);
            dst[0  + n] = o0[r] * inv_l;
            dst[16 + n] = o1[r] * inv_l;
            dst[32 + n] = o2[r] * inv_l;
            dst[48 + n] = o3[r] * inv_l;
        }
    }
}

extern "C" void kernel_launch(void* const* d_in, const int* in_sizes, int n_in,
                              void* d_out, int out_size, void* d_ws, size_t ws_size,
                              hipStream_t stream) {
    const float* q = (const float*)d_in[0];
    const float* k = (const float*)d_in[1];
    const float* v = (const float*)d_in[2];
    float* out = (float*)d_out;

    // workspace layout
    char* ws = (char*)d_ws;
    int* perm = (int*)ws;                                   // 4096 ints
    int* gidx = perm + NSEQ;                                // 3*2048 ints
    const size_t SLOT_ELEMS = (size_t)2048 * DH;            // per-slot elems
    _Float16* Qg  = (_Float16*)(ws + 65536);
    _Float16* Ktg = Qg  + NSLOT * SLOT_ELEMS;
    _Float16* Vg  = Ktg + NSLOT * SLOT_ELEMS;

    hilbert_perm_kernel<<<NSEQ / 256, 256, 0, stream>>>(perm);
    gather_idx_kernel<<<2048 / 256, 256, 0, stream>>>(perm, gidx);
    gather_cvt_kernel<<<dim3(2048 / 256, NSLOT), 256, 0, stream>>>(
        q, k, v, gidx, Qg, Ktg, Vg);

    int n4 = out_size / 4;
    zero_out_kernel<<<(n4 + 255) / 256, 256, 0, stream>>>((float4*)d_out, n4);

    dim3 grid(2048 / BQ, 3, 2 * 8);
    size_t smem_bytes = (size_t)(BQ * STR + 4 * BK * STR + 8 * 16 * STR) * sizeof(_Float16);
    dilated_flash_attn_kernel<<<grid, 256, smem_bytes, stream>>>(Qg, Ktg, Vg, gidx, out);
}
